// QuantMLPBlock_74122545594496
// MI455X (gfx1250) — 8-bit matrix operand order repaired and checked by reading, compiled, not run on hardware
//
#include <hip/hip_runtime.h>
#include <hip/hip_fp16.h>
#include <math.h>
#include <stdint.h>


typedef __attribute__((ext_vector_type(8))) int v8i;

static constexpr int  BT = 8 * 4096;
static constexpr int  DD = 768;
static constexpr int  HH = 3072;

static constexpr size_t OFF_AMAX = 0;
static constexpr size_t OFF_XQ   = 256;
static constexpr size_t SZ_XQ    = (size_t)BT * DD;
static constexpr size_t OFF_W1Q  = OFF_XQ + SZ_XQ;
static constexpr size_t SZ_W1Q   = (size_t)HH * DD;
static constexpr size_t OFF_W2Q  = OFF_W1Q + SZ_W1Q;
static constexpr size_t SZ_W2Q   = (size_t)DD * HH;
static constexpr size_t OFF_H    = OFF_W2Q + SZ_W2Q;
static constexpr size_t SZ_H     = (size_t)BT * HH * 2;
static constexpr size_t OFF_HQ   = OFF_H + SZ_H;


__device__ __forceinline__ float scale_from_amax(const unsigned* slot) {
  return fmaxf(__uint_as_float(*slot) * (1.0f / 127.0f), 1e-8f);
}

__device__ __forceinline__ void wave_reduce_max_atomic(float am, unsigned* out) {
  #pragma unroll
  for (int off = 16; off > 0; off >>= 1)
    am = fmaxf(am, __shfl_xor(am, off, 32));
  if ((threadIdx.x & 31) == 0)
    atomicMax(out, __float_as_uint(am));
}

__global__ __launch_bounds__(256) void absmax_f32_kernel(
    const float4* __restrict__ x, long n4, unsigned* __restrict__ out) {
  long i      = (long)blockIdx.x * blockDim.x + threadIdx.x;
  long stride = (long)gridDim.x * blockDim.x;
  float am = 0.0f;
  for (; i < n4; i += stride) {
    float4 v = x[i];
    am = fmaxf(am, fmaxf(fmaxf(fabsf(v.x), fabsf(v.y)),
                         fmaxf(fabsf(v.z), fabsf(v.w))));
  }
  wave_reduce_max_atomic(am, out);
}

__global__ __launch_bounds__(256) void quant_f32_kernel(
    const float4* __restrict__ x, char4* __restrict__ q, long n4,
    const unsigned* __restrict__ amax_slot, float qmin, float qmax) {
  const float inv_s = 1.0f / scale_from_amax(amax_slot);
  long i      = (long)blockIdx.x * blockDim.x + threadIdx.x;
  long stride = (long)gridDim.x * blockDim.x;
  for (; i < n4; i += stride) {
    float4 v = x[i];
    char4 c;
    c.x = (signed char)fminf(fmaxf(rintf(v.x * inv_s), qmin), qmax);
    c.y = (signed char)fminf(fmaxf(rintf(v.y * inv_s), qmin), qmax);
    c.z = (signed char)fminf(fmaxf(rintf(v.z * inv_s), qmin), qmax);
    c.w = (signed char)fminf(fmaxf(rintf(v.w * inv_s), qmin), qmax);
    q[i] = c;
  }
}

__global__ __launch_bounds__(256) void quant_f16_kernel(
    const __half2* __restrict__ x, short* __restrict__ q, long n2,
    const unsigned* __restrict__ amax_slot, float qmin, float qmax) {
  const float inv_s = 1.0f / scale_from_amax(amax_slot);
  long i      = (long)blockIdx.x * blockDim.x + threadIdx.x;
  long stride = (long)gridDim.x * blockDim.x;
  for (; i < n2; i += stride) {
    __half2 hv = x[i];
    float a = __low2float(hv), b = __high2float(hv);
    int qa = (int)fminf(fmaxf(rintf(a * inv_s), qmin), qmax);
    int qb = (int)fminf(fmaxf(rintf(b * inv_s), qmin), qmax);
    q[i] = (short)((qa & 0xff) | (qb << 8));
  }
}

__device__ __forceinline__ v8i load_a_frag(const signed char* __restrict__ p) {
  v8i a;
  #pragma unroll
  for (int c = 0; c < 4; ++c) {
    int2 t = *(const int2*)(p + c * 16);
    a[2 * c]     = t.x;
    a[2 * c + 1] = t.y;
  }
  return a;
}

__device__ __forceinline__ v8i load_b_frag(const signed char* __restrict__ p) {
  v8i b;
  #pragma unroll
  for (int c = 0; c < 4; ++c) {
    int2 t = *(const int2*)(p + c * 16);
    b[2 * c]     = t.x;
    b[2 * c + 1] = t.y;
  }
  return b;
}

template <int K>
__device__ __forceinline__ void gemm_i8_core(const signed char* __restrict__ A,
                                             const signed char* __restrict__ B,
                                             int m_base, int n_base, int l, int g,
                                             v8i acc[8]) {
  const signed char* pa0 = A + (long)(m_base + l)      * K + g * 8;
  const signed char* pa1 = A + (long)(m_base + 16 + l) * K + g * 8;
  const signed char* pb0 = B + (long)(n_base + l)      * K + g * 8;
  const signed char* pb1 = pb0 + (long)16 * K;
  const signed char* pb2 = pb0 + (long)32 * K;
  const signed char* pb3 = pb0 + (long)48 * K;

  for (int k0 = 0; k0 < K; k0 += 256) {
    if (k0 + 256 < K) {
      __builtin_prefetch(pa0 + 256, 0, 3);
      __builtin_prefetch(pa1 + 256, 0, 3);
    }
    #pragma unroll
    for (int c = 0; c < 4; ++c) {
      const int o = c * 64;
      v8i a0 = load_a_frag(pa0 + o);
      v8i a1 = load_a_frag(pa1 + o);
      v8i b0 = load_b_frag(pb0 + o);
      v8i b1 = load_b_frag(pb1 + o);
      v8i b2 = load_b_frag(pb2 + o);
      v8i b3 = load_b_frag(pb3 + o);
      acc[0] = __builtin_amdgcn_wmma_i32_16x16x64_iu8(true, a0, true, b0, acc[0], false, false);
      acc[1] = __builtin_amdgcn_wmma_i32_16x16x64_iu8(true, a0, true, b1, acc[1], false, false);
      acc[2] = __builtin_amdgcn_wmma_i32_16x16x64_iu8(true, a0, true, b2, acc[2], false, false);
      acc[3] = __builtin_amdgcn_wmma_i32_16x16x64_iu8(true, a0, true, b3, acc[3], false, false);
      acc[4] = __builtin_amdgcn_wmma_i32_16x16x64_iu8(true, a1, true, b0, acc[4], false, false);
      acc[5] = __builtin_amdgcn_wmma_i32_16x16x64_iu8(true, a1, true, b1, acc[5], false, false);
      acc[6] = __builtin_amdgcn_wmma_i32_16x16x64_iu8(true, a1, true, b2, acc[6], false, false);
      acc[7] = __builtin_amdgcn_wmma_i32_16x16x64_iu8(true, a1, true, b3, acc[7], false, false);
    }
    pa0 += 256; pa1 += 256;
    pb0 += 256; pb1 += 256; pb2 += 256; pb3 += 256;
  }
}

__device__ __forceinline__ void tile_coords(int& m_base, int& n_base, int& l, int& g) {
  const int lane = threadIdx.x & 31;
  const int wid  = threadIdx.x >> 5;
  l = lane & 15;
  g = lane >> 4;
  m_base = blockIdx.y * 128 + (wid >> 1) * 32;
  n_base = blockIdx.x * 128 + (wid & 1) * 64;
}

__global__ __launch_bounds__(256) void gemm1_kernel(
    const signed char* __restrict__ A,
    const signed char* __restrict__ B,
    const float* __restrict__ bias,
    const unsigned* __restrict__ amax,
    __half* __restrict__ Hout,
    unsigned* __restrict__ amaxH) {
  int m_base, n_base, l, g;
  tile_coords(m_base, n_base, l, g);

  v8i acc[8] = {};
  gemm_i8_core<DD>(A, B, m_base, n_base, l, g, acc);

  const float s = scale_from_amax(&amax[0]) * scale_from_amax(&amax[1]);
  float am = 0.0f;

  #pragma unroll
  for (int tm = 0; tm < 2; ++tm) {
    #pragma unroll
    for (int tn = 0; tn < 4; ++tn) {
      const v8i& a   = acc[tm * 4 + tn];
      const int n    = n_base + tn * 16 + l;
      const float bn = bias[n];
      const int mrow = m_base + tm * 16 + g * 8;
      #pragma unroll
      for (int rr = 0; rr < 8; ++rr) {
        float v  = (float)a[rr] * s + bn;
        float ge = 0.5f * v * (1.0f + erff(v * 0.70710678118654752f));
        Hout[(long)(mrow + rr) * HH + n] = __float2half(ge);
        am = fmaxf(am, fabsf(ge));
      }
    }
  }
  wave_reduce_max_atomic(am, amaxH);
}

__global__ __launch_bounds__(256) void gemm2_kernel(
    const signed char* __restrict__ A,
    const signed char* __restrict__ B,
    const float* __restrict__ bias,
    const unsigned* __restrict__ amax,
    float* __restrict__ Out) {
  int m_base, n_base, l, g;
  tile_coords(m_base, n_base, l, g);

  v8i acc[8] = {};
  gemm_i8_core<HH>(A, B, m_base, n_base, l, g, acc);

  const float s = scale_from_amax(&amax[2]) * scale_from_amax(&amax[3]);

  #pragma unroll
  for (int tm = 0; tm < 2; ++tm) {
    #pragma unroll
    for (int tn = 0; tn < 4; ++tn) {
      const v8i& a   = acc[tm * 4 + tn];
      const int n    = n_base + tn * 16 + l;
      const float bn = bias[n];
      const int mrow = m_base + tm * 16 + g * 8;
      #pragma unroll
      for (int rr = 0; rr < 8; ++rr) {
        Out[(long)(mrow + rr) * DD + n] = (float)a[rr] * s + bn;
      }
    }
  }
}

extern "C" void kernel_launch(void* const* d_in, const int* in_sizes, int n_in,
                              void* d_out, int out_size, void* d_ws, size_t ws_size,
                              hipStream_t stream) {
  (void)in_sizes; (void)n_in; (void)out_size; (void)ws_size;

  const float* x  = (const float*)d_in[0];
  const float* W1 = (const float*)d_in[1];
  const float* b1 = (const float*)d_in[2];
  const float* W2 = (const float*)d_in[3];
  const float* b2 = (const float*)d_in[4];
  float* out = (float*)d_out;

  char* ws = (char*)d_ws;
  unsigned*    amax = (unsigned*)(ws + OFF_AMAX);
  signed char* xq   = (signed char*)(ws + OFF_XQ);
  signed char* w1q  = (signed char*)(ws + OFF_W1Q);
  signed char* w2q  = (signed char*)(ws + OFF_W2Q);
  __half*      h    = (__half*)(ws + OFF_H);
  signed char* hq   = (signed char*)(ws + OFF_HQ);

  hipMemsetAsync(ws, 0, 64, stream);

  const long nX  = (long)BT * DD;
  const long nW1 = (long)HH * DD;
  const long nW2 = (long)DD * HH;
  const long nH  = (long)BT * HH;

  absmax_f32_kernel<<<8192, 256, 0, stream>>>((const float4*)x,  nX  / 4, &amax[0]);
  absmax_f32_kernel<<<2048, 256, 0, stream>>>((const float4*)W1, nW1 / 4, &amax[1]);
  absmax_f32_kernel<<<2048, 256, 0, stream>>>((const float4*)W2, nW2 / 4, &amax[3]);

  quant_f32_kernel<<<8192, 256, 0, stream>>>((const float4*)x,  (char4*)xq,  nX  / 4, &amax[0], -128.0f, 127.0f);
  quant_f32_kernel<<<2048, 256, 0, stream>>>((const float4*)W1, (char4*)w1q, nW1 / 4, &amax[1], -127.0f, 127.0f);
  quant_f32_kernel<<<2048, 256, 0, stream>>>((const float4*)W2, (char4*)w2q, nW2 / 4, &amax[3], -127.0f, 127.0f);

  dim3 g1(HH / 128, BT / 128);
  gemm1_kernel<<<g1, 256, 0, stream>>>(xq, w1q, b1, amax, h, &amax[2]);

  quant_f16_kernel<<<16384, 256, 0, stream>>>((const __half2*)h, (short*)hq, nH / 2, &amax[2], -128.0f, 127.0f);

  dim3 g2(DD / 128, BT / 128);
  gemm2_kernel<<<g2, 256, 0, stream>>>(hq, w2q, b2, amax, out);
}
